// Colorizer_63917703299822
// MI455X (gfx1250) — matrix operand order repaired and checked by reading, not compiled, not run on hardware
//
#include <hip/hip_runtime.h>

// ---------------- constants (match reference) ----------------
#define H64   64
#define W64   64
#define CCH   32          // feature channels == WMMA K
#define NPOS  4096        // 64*64
#define NREF  4
#define PMW   25          // long-range window
#define RMW   12
#define P13   13          // short-range window
#define R6    6
#define N169  169
#define CCLS  32
#define BETA  50.0f

typedef _Float16 half16 __attribute__((ext_vector_type(16)));
typedef float   float8 __attribute__((ext_vector_type(8)));

// ---------------- workspace layout (bytes) ----------------
// flags[4] | dirate[4] | misc[2] | offset f32[4][4096][2] | grid f32[4][4096][2]
// qds i32[4][4096] | tH f16[64][64][32] | rH f16[4][64][64][32]
// corr_pm f32[625][4096] | corrS f32[169][4096]
#define OFF_FLAGS   0
#define OFF_DIRATE  16
#define OFF_MISC    32
#define OFF_OFFSET  256
#define OFF_GRID    (OFF_OFFSET + 4*NPOS*2*4)          // 131328
#define OFF_QDS     (OFF_GRID   + 4*NPOS*2*4)          // 262400
#define OFF_TH      (OFF_QDS    + 4*NPOS*4)            // 327936
#define OFF_RH      (OFF_TH     + NPOS*CCH*2)          // 590080
#define OFF_CORRPM  (OFF_RH     + NREF*NPOS*CCH*2)     // 1638656
#define OFF_CORRS   (OFF_CORRPM + PMW*PMW*NPOS*4)      // 11878656

// ---------------- prep: device-side config ----------------
__global__ void prep_kernel(const int* ref_index, const int* cur_ind,
                            int* flags, int* dirate, int* misc, float* out_scalar) {
    if (threadIdx.x == 0) {
        int ci = cur_ind[0];
        int last = -1, nshort = 0;
        for (int i = 0; i < NREF; ++i) {
            int d = ci - ref_index[i];
            int s = (d > 15) ? 1 : 0;
            flags[i] = s;
            int dr = d / 15 + 1; if (dr > 4) dr = 4;
            dirate[i] = s ? dr : 1;
            if (s) last = i; else nshort++;
        }
        misc[0] = last;
        misc[1] = nshort;
        out_scalar[0] = 0.0f;   // smoothness accumulator
    }
}

// ---------------- f32 -> f16 staging in [y][x][c] order ----------------
__global__ void convert_kernel(const float* fr, const float* ft,
                               _Float16* tH, _Float16* rH) {
    int idx = blockIdx.x * blockDim.x + threadIdx.x;
    if (idx < CCH * NPOS) {
        int c = idx / NPOS, pos = idx % NPOS;
        tH[pos * CCH + c] = (_Float16)ft[idx];
    }
    if (idx < NREF * CCH * NPOS) {
        int ref = idx / (CCH * NPOS);
        int rem = idx % (CCH * NPOS);
        int c = rem / NPOS, pos = rem % NPOS;
        rH[(ref * NPOS + pos) * CCH + c] = (_Float16)fr[idx];
    }
}

// ---------------- label downsample (::4,::4) ----------------
__global__ void qds_kernel(const int* qr, int* qds) {
    int idx = blockIdx.x * blockDim.x + threadIdx.x;
    if (idx >= NREF * NPOS) return;
    int ref = idx / NPOS, pos = idx % NPOS;
    int y = pos / W64, x = pos % W64;
    qds[idx] = qr[ref * 256 * 256 + (y * 4) * 256 + (x * 4)];
}

// ---------------- windowed correlation via WMMA row-pair GEMM ----------------
// One workgroup (64 thr = 2 waves) per (p, y). Computes RowCorr[x][xs] =
// sum_c tH[y][x][c]*rH[ys][xs][c] with 16 x v_wmma_f32_16x16x32_f16 (K=32 = full
// channel reduction), then scatters the dilated q-diagonals to corr_out.
__global__ void __launch_bounds__(64)
corr_win_kernel(const _Float16* __restrict__ tH, const _Float16* __restrict__ rH,
                const int* flags, const int* dirate, float* corr_out,
                int ref, int Pp, int Rp, int useDirate, int wantSearch) {
    if (flags[ref] != wantSearch) return;
    int dil = useDirate ? dirate[ref] : 1;
    int p = blockIdx.x;            // window row shift
    int y = blockIdx.y;            // target row
    int ys = y + (p - Rp) * dil;
    int tid = threadIdx.x;

    if (ys < 0 || ys >= H64) {     // fully padded row -> zeros
        for (int i = tid; i < Pp * W64; i += 64) {
            int q = i >> 6, x = i & 63;
            corr_out[(p * Pp + q) * NPOS + y * W64 + x] = 0.0f;
        }
        return;
    }

    __shared__ float rowc[64][65];
    int wave = tid >> 5, lane = tid & 31;
    int l15 = lane & 15, hi = lane >> 4;
    const _Float16* trow = tH + y * W64 * CCH;
    const _Float16* srow = rH + (ref * NPOS + ys * W64) * CCH;

    for (int mi = 0; mi < 2; ++mi) {
        int mt = wave * 2 + mi;
        // A fragment: 16x32 f16 — lane holds row M=l15, K = hi*8 + {0..7, 16..23}
        half16 a;
        {
            const _Float16* tp = trow + (mt * 16 + l15) * CCH + hi * 8;
            #pragma unroll
            for (int e = 0; e < 8; ++e) a[e] = tp[e];
            #pragma unroll
            for (int e = 0; e < 8; ++e) a[8 + e] = tp[16 + e];
        }
        #pragma unroll
        for (int nt = 0; nt < 4; ++nt) {
            // B fragment: 32x16 f16 — lane holds col N=l15, K = hi*8 + {0..7, 16..23}
            half16 b;
            {
                const _Float16* sp = srow + (nt * 16 + l15) * CCH + hi * 8;
                #pragma unroll
                for (int e = 0; e < 8; ++e) b[e] = sp[e];
                #pragma unroll
                for (int e = 0; e < 8; ++e) b[8 + e] = sp[16 + e];
            }
            float8 c = {};
            c = __builtin_amdgcn_wmma_f32_16x16x32_f16(
                    false, a, false, b, (short)0, c, false, false);
            // C layout: VGPR v -> M = hi*8 + v, N = l15
            #pragma unroll
            for (int v = 0; v < 8; ++v)
                rowc[mt * 16 + hi * 8 + v][nt * 16 + l15] = c[v];
        }
    }
    __syncthreads();

    for (int i = tid; i < Pp * W64; i += 64) {
        int q = i >> 6, x = i & 63;
        int xs = x + (q - Rp) * dil;
        float v = (xs >= 0 && xs < W64) ? rowc[x][xs] : 0.0f;
        corr_out[(p * Pp + q) * NPOS + y * W64 + x] = v;
    }
}

// ---------------- plain softmax + soft-argmax over 25x25 -> offsets ----------------
__global__ void offset_kernel(const float* corr, const int* flags, const int* dirate,
                              float* offset, int ref) {
    if (flags[ref] != 1) return;
    int pos = blockIdx.x * blockDim.x + threadIdx.x;
    if (pos >= NPOS) return;
    float m = -1e30f;
    for (int i = 0; i < PMW * PMW; ++i) m = fmaxf(m, corr[i * NPOS + pos]);
    float s = 0.f, sx = 0.f, sy = 0.f;
    for (int i = 0; i < PMW * PMW; ++i) {
        float e = __expf(corr[i * NPOS + pos] - m);
        s  += e;
        sx += e * (float)((i % PMW) - RMW);
        sy += e * (float)((i / PMW) - RMW);
    }
    float dr = (float)dirate[ref];
    offset[(ref * NPOS + pos) * 2 + 0] = sx / s * dr;
    offset[(ref * NPOS + pos) * 2 + 1] = sy / s * dr;
}

// ---------------- deform-patch correlation (off-grid bilinear; VALU) ----------------
__global__ void corr_deform_kernel(const float* __restrict__ feats_r,
                                   const float* __restrict__ feats_t,
                                   const float* offset, const int* flags,
                                   float* corrS, int ref) {
    if (flags[ref] != 1) return;
    int idx = blockIdx.x * blockDim.x + threadIdx.x;
    if (idx >= N169 * NPOS) return;
    int n = idx / NPOS, pos = idx % NPOS;
    int y = pos / W64, x = pos % W64;
    float offx = offset[(ref * NPOS + pos) * 2 + 0];
    float offy = offset[(ref * NPOS + pos) * 2 + 1];
    float py = offy + (float)y + (float)(n / P13 - R6);
    float px = offx + (float)x + (float)(n % P13 - R6);
    float y0f = floorf(py), x0f = floorf(px);
    int y0 = (int)y0f, x0 = (int)x0f;
    float wy = py - y0f, wx = px - x0f;
    float w00 = (1.f - wy) * (1.f - wx), w01 = (1.f - wy) * wx;
    float w10 = wy * (1.f - wx),          w11 = wy * wx;
    bool vy0 = (y0 >= 0 && y0 < H64), vy1 = (y0 + 1 >= 0 && y0 + 1 < H64);
    bool vx0 = (x0 >= 0 && x0 < W64), vx1 = (x0 + 1 >= 0 && x0 + 1 < W64);
    const float* fr = feats_r + ref * CCH * NPOS;
    float acc = 0.f;
    for (int c = 0; c < CCH; ++c) {
        const float* f = fr + c * NPOS;
        float v00 = (vy0 && vx0) ? f[y0 * W64 + x0]           : 0.f;
        float v01 = (vy0 && vx1) ? f[y0 * W64 + x0 + 1]       : 0.f;
        float v10 = (vy1 && vx0) ? f[(y0 + 1) * W64 + x0]     : 0.f;
        float v11 = (vy1 && vx1) ? f[(y0 + 1) * W64 + x0 + 1] : 0.f;
        acc += feats_t[c * NPOS + pos] * (w00 * v00 + w01 * v01 + w10 * v10 + w11 * v11);
    }
    corrS[n * NPOS + pos] = acc;
}

// ---------------- beta-softmax soft-argmax over 13x13 -> grid ----------------
__global__ void grid_kernel(const float* corrS, float* grid, int ref) {
    int pos = blockIdx.x * blockDim.x + threadIdx.x;
    if (pos >= NPOS) return;
    float m = -1e30f;
    for (int n = 0; n < N169; ++n) m = fmaxf(m, corrS[n * NPOS + pos]);
    float s = 0.f, gx = 0.f, gy = 0.f;
    for (int n = 0; n < N169; ++n) {
        float e = __expf(BETA * (corrS[n * NPOS + pos] - m));
        s  += e;
        gx += e * (-1.f + (1.f / 6.f) * (float)(n % P13));
        gy += e * (-1.f + (1.f / 6.f) * (float)(n / P13));
    }
    grid[(ref * NPOS + pos) * 2 + 0] = gx / s;
    grid[(ref * NPOS + pos) * 2 + 1] = gy / s;
}

// ---------------- output: one-hot-free grid_sample + channel regroup ----------------
// out[a,y,x], a = ref*8 + (cls*169+n)/676 ; grid_sample weight goes straight to the
// label's channel group. Deform refs use offset of the LAST search ref (reference quirk).
__global__ void output_kernel(const int* __restrict__ qds, const float* grid,
                              const float* offset, const int* flags, const int* misc,
                              float* out) {
    int idx = blockIdx.x * blockDim.x + threadIdx.x;
    if (idx >= NREF * NPOS) return;
    int ref = idx / NPOS, pos = idx % NPOS;
    float gx = grid[(ref * NPOS + pos) * 2 + 0];
    float gy = grid[(ref * NPOS + pos) * 2 + 1];
    float px = ((gx + 1.f) * (float)W64 - 1.f) * 0.5f;
    float py = ((gy + 1.f) * (float)H64 - 1.f) * 0.5f;
    float x0f = floorf(px), y0f = floorf(py);
    int x0 = (int)x0f, y0 = (int)y0f;
    float wx = px - x0f, wy = py - y0f;
    int isS = flags[ref];
    int last = misc[0];
    const int* lab = qds + ref * NPOS;
    float acc[8] = {0.f, 0.f, 0.f, 0.f, 0.f, 0.f, 0.f, 0.f};

    for (int corner = 0; corner < 4; ++corner) {
        int ys = y0 + (corner >> 1), xs = x0 + (corner & 1);
        float w = ((corner >> 1) ? wy : 1.f - wy) * ((corner & 1) ? wx : 1.f - wx);
        if (ys < 0 || ys >= H64 || xs < 0 || xs >= W64) continue;
        if (!isS) {
            for (int n = 0; n < N169; ++n) {
                int yy = ys + n / P13 - R6, xx = xs + n % P13 - R6;
                if (yy < 0 || yy >= H64 || xx < 0 || xx >= W64) continue;
                int cls = lab[yy * W64 + xx];
                acc[(unsigned)(cls * N169 + n) / 676u] += w;
            }
        } else {
            float ox = offset[(last * NPOS + ys * W64 + xs) * 2 + 0];
            float oy = offset[(last * NPOS + ys * W64 + xs) * 2 + 1];
            for (int n = 0; n < N169; ++n) {
                float py2 = oy + (float)(ys + n / P13 - R6);
                float px2 = ox + (float)(xs + n % P13 - R6);
                float yb = floorf(py2), xb = floorf(px2);
                int yi = (int)yb, xi = (int)xb;
                float fy = py2 - yb, fx = px2 - xb;
                #pragma unroll
                for (int c2 = 0; c2 < 4; ++c2) {
                    int yy = yi + (c2 >> 1), xx = xi + (c2 & 1);
                    if (yy < 0 || yy >= H64 || xx < 0 || xx >= W64) continue;
                    float w2 = ((c2 >> 1) ? fy : 1.f - fy) * ((c2 & 1) ? fx : 1.f - fx);
                    int cls = lab[yy * W64 + xx];
                    acc[(unsigned)(cls * N169 + n) / 676u] += w * w2;
                }
            }
        }
    }
    #pragma unroll
    for (int j = 0; j < 8; ++j)
        out[(ref * 8 + j) * NPOS + pos] = acc[j];
}

// ---------------- smoothness scalar ----------------
__global__ void smooth_kernel(const float* grid, const int* flags, const int* misc,
                              float* out_scalar) {
    __shared__ float red[256];
    int idx = blockIdx.x * blockDim.x + threadIdx.x;
    float val = 0.f;
    if (idx < NREF * NPOS) {
        int ref = idx / NPOS, pos = idx % NPOS;
        if (!flags[ref]) {
            int y = pos / W64, x = pos % W64;
            auto F = [&](int yy, int xx) -> float {
                if (yy < 0 || yy >= H64 || xx < 0 || xx >= W64) return 0.f;
                float g = grid[(ref * NPOS + yy * W64 + xx) * 2 + 0];
                float X = -1.f + 2.f * (float)xx / 63.f;
                float Y = -1.f + 2.f * (float)yy / 63.f;
                return 2.f * g - X - Y;   // sum over the two flow channels
            };
            float dxv = (F(y, x + 1) - F(y, x - 1)) * 0.5f;
            float dyv = (F(y + 1, x) - F(y - 1, x)) * 0.5f;
            val = fabsf(dxv) + fabsf(dyv);
        }
    }
    red[threadIdx.x] = val;
    __syncthreads();
    for (int s = 128; s > 0; s >>= 1) {
        if (threadIdx.x < s) red[threadIdx.x] += red[threadIdx.x + s];
        __syncthreads();
    }
    if (threadIdx.x == 0) {
        int nshort = misc[1];
        float denom = (float)(nshort * 2 * NPOS);
        atomicAdd(out_scalar, denom > 0.f ? red[0] / denom : 0.f);
    }
}

// ---------------- host orchestration ----------------
extern "C" void kernel_launch(void* const* d_in, const int* in_sizes, int n_in,
                              void* d_out, int out_size, void* d_ws, size_t ws_size,
                              hipStream_t stream) {
    const float* feats_r     = (const float*)d_in[0];
    const float* feats_t     = (const float*)d_in[1];
    const int*   quantized_r = (const int*)d_in[2];
    const int*   ref_index   = (const int*)d_in[3];
    const int*   current_ind = (const int*)d_in[4];
    float* out = (float*)d_out;
    char*  ws  = (char*)d_ws;

    int*      flags   = (int*)(ws + OFF_FLAGS);
    int*      dirate  = (int*)(ws + OFF_DIRATE);
    int*      misc    = (int*)(ws + OFF_MISC);
    float*    offset  = (float*)(ws + OFF_OFFSET);
    float*    grid    = (float*)(ws + OFF_GRID);
    int*      qds     = (int*)(ws + OFF_QDS);
    _Float16* tH      = (_Float16*)(ws + OFF_TH);
    _Float16* rH      = (_Float16*)(ws + OFF_RH);
    float*    corr_pm = (float*)(ws + OFF_CORRPM);
    float*    corrS   = (float*)(ws + OFF_CORRS);

    prep_kernel<<<1, 64, 0, stream>>>(ref_index, current_ind, flags, dirate, misc,
                                      out + CCLS * NPOS);
    convert_kernel<<<(NREF * CCH * NPOS + 255) / 256, 256, 0, stream>>>(
        feats_r, feats_t, tH, rH);
    qds_kernel<<<(NREF * NPOS + 255) / 256, 256, 0, stream>>>(quantized_r, qds);

    for (int ref = 0; ref < NREF; ++ref) {
        // long-range path (search refs): 25x25 dilated correlation via WMMA
        corr_win_kernel<<<dim3(PMW, H64), 64, 0, stream>>>(
            tH, rH, flags, dirate, corr_pm, ref, PMW, RMW, /*useDirate=*/1, /*want=*/1);
        offset_kernel<<<(NPOS + 255) / 256, 256, 0, stream>>>(
            corr_pm, flags, dirate, offset, ref);
        corr_deform_kernel<<<(N169 * NPOS + 255) / 256, 256, 0, stream>>>(
            feats_r, feats_t, offset, flags, corrS, ref);
        // short-range path (non-search refs): 13x13 correlation via WMMA
        corr_win_kernel<<<dim3(P13, H64), 64, 0, stream>>>(
            tH, rH, flags, dirate, corrS, ref, P13, R6, /*useDirate=*/0, /*want=*/0);
        grid_kernel<<<(NPOS + 255) / 256, 256, 0, stream>>>(corrS, grid, ref);
    }

    output_kernel<<<(NREF * NPOS + 127) / 128, 128, 0, stream>>>(
        qds, grid, offset, flags, misc, out);
    smooth_kernel<<<(NREF * NPOS + 255) / 256, 256, 0, stream>>>(
        grid, flags, misc, out + CCLS * NPOS);
}
